// MMG_single_33663953666654
// MI455X (gfx1250) — matrix operand order repaired and checked by reading, not compiled, not run on hardware
//
#include <hip/hip_runtime.h>
#include <hip/hip_bf16.h>
#include <math.h>
#include <stdint.h>

#define N_NODES 2048
#define N_EDGES 32768
#define DIM 256
#define HEADS 4

typedef __bf16 bf16_t;
typedef __attribute__((ext_vector_type(16))) __bf16 v16bf;
typedef __attribute__((ext_vector_type(8)))  float  v8f;

static __device__ __forceinline__ v8f wmma_bf16(v16bf a, v16bf b, v8f c) {
  // v_wmma_f32_16x16x32_bf16: D = A(16x32) * B(32x16) + C(16x16 f32)
  return __builtin_amdgcn_wmma_f32_16x16x32_bf16(false, a, false, b, (short)0, c, false, false);
}

// async global -> LDS copy of 16 bytes (per-lane gather), tracked by ASYNCcnt
static __device__ __forceinline__ void async_b128(const void* gp, const void* lds_ptr) {
  unsigned lds = (unsigned)(uintptr_t)lds_ptr;              // low 32 bits == LDS offset
  unsigned long long ga = (unsigned long long)(uintptr_t)gp;
  asm volatile("global_load_async_to_lds_b128 %0, %1, off" :: "v"(lds), "v"(ga) : "memory");
}
static __device__ __forceinline__ void wait_async0() {
  asm volatile("s_wait_asynccnt 0" ::: "memory");
}

// float atomic max via monotonic int trick (init must be -inf)
static __device__ __forceinline__ void atomicMaxF(float* addr, float v) {
  if (v >= 0.0f) atomicMax((int*)addr, __float_as_int(v));
  else           atomicMin((unsigned int*)addr, __float_as_uint(v));
}

// ---------------------------------------------------------------------------
// Tiled GEMM on bf16 operands:
//   C[M,N] = act( (accum ? Cf : 0) + A[M,K](row-gather) * W[K,N] + bias )
// A,W bf16 in global. A tile staged via global_load_async_to_lds_b128
// (per-lane gather; idx<0 redirected to a zero page). W tile staged with
// b128 loads + transposed LDS stores for pair-contiguous B-frag reads.
// fp32 accumulate; writes fp32 (writeF) and/or bf16 (Cb) outputs.
// M%64==0, N%64==0, K%32==0.
// ---------------------------------------------------------------------------
__global__ __launch_bounds__(128) void gemm_k(
    const bf16_t* __restrict__ A, int lda, const int* __restrict__ aidx,
    const bf16_t* __restrict__ zp,
    const bf16_t* __restrict__ W, int ldw,
    const float* __restrict__ bias,
    int accum, int writeF, int act, float oscale,
    float* __restrict__ Cf, bf16_t* __restrict__ Cb, int ldc,
    int M, int N, int K)
{
  __shared__ bf16_t As[64][40];   // 80B row stride (16B aligned for async b128)
  __shared__ bf16_t Ws[64][34];   // transposed [col][k]
  const int nb = N >> 6;
  const int bm = blockIdx.x / nb, bn = blockIdx.x % nb;
  const int m0 = bm << 6, n0 = bn << 6;
  const int t = threadIdx.x;
  const int lane = t & 31, wave = t >> 5;
  const int hi = lane >> 4, ln = lane & 15;

  v8f c[4];
#pragma unroll
  for (int i = 0; i < 4; i++)
#pragma unroll
    for (int j = 0; j < 8; j++) c[i][j] = 0.0f;

  for (int k0 = 0; k0 < K; k0 += 32) {
    // --- stage A tile (64 rows x 32 cols bf16) via async gather copies ---
#pragma unroll
    for (int i = 0; i < 2; i++) {
      int chunk = i * 128 + t;            // 256 chunks of 16B
      int row = chunk >> 2, c16 = chunk & 3;
      int gr = m0 + row;
      int srow = aidx ? aidx[gr] : gr;
      const bf16_t* gp = (srow >= 0) ? (A + (long)srow * lda + k0 + c16 * 8)
                                     : (zp + c16 * 8);
      async_b128(gp, &As[row][c16 * 8]);
    }
    // --- stage W tile (32 k-rows x 64 cols), transpose into Ws[col][k] ---
#pragma unroll
    for (int i = 0; i < 2; i++) {
      int f8 = i * 128 + t;               // 256 vec8 chunks
      int kk = f8 >> 3, n8 = f8 & 7;
      uint4 d = *reinterpret_cast<const uint4*>(W + (long)(k0 + kk) * ldw + n0 + n8 * 8);
      const bf16_t* c8 = (const bf16_t*)&d;
#pragma unroll
      for (int z = 0; z < 8; z++) Ws[n8 * 8 + z][kk] = c8[z];
    }
    wait_async0();
    __syncthreads();

    // A fragment: this wave's 16 rows, K=32
    v16bf af;
    const int arow = (wave << 4) + ln;
#pragma unroll
    for (int e = 0; e < 16; e++) {
      int j = e >> 1, p = e & 1;
      int kk = ((j < 4) ? 0 : 16) + hi * 8 + (j & 3) * 2 + p;
      af[e] = As[arow][kk];
    }
#pragma unroll
    for (int ct = 0; ct < 4; ct++) {
      v16bf bfv;
      const int bcol = (ct << 4) + ln;
#pragma unroll
      for (int e = 0; e < 16; e++) {
        int j = e >> 1, p = e & 1;
        int kk = ((j < 4) ? 0 : 16) + hi * 8 + (j & 3) * 2 + p;
        bfv[e] = Ws[bcol][kk];
      }
      c[ct] = wmma_bf16(af, bfv, c[ct]);
    }
    __syncthreads();
  }

  // epilogue
#pragma unroll
  for (int ct = 0; ct < 4; ct++) {
    int gcol = n0 + (ct << 4) + ln;
    float b = bias ? bias[gcol] : 0.0f;
#pragma unroll
    for (int j = 0; j < 8; j++) {
      int grow = m0 + (wave << 4) + j + 8 * hi;
      long off = (long)grow * ldc + gcol;
      float v = c[ct][j];
      if (accum) v += Cf[off];
      v += b;
      v *= oscale;
      if (act == 1) v = fmaxf(v, 0.0f);
      else if (act == 2) v = 1.0f / (1.0f + __expf(-v));
      if (writeF) Cf[off] = v;
      if (Cb)     Cb[off] = (bf16_t)v;
    }
  }
}

// ---------------------------------------------------------------------------
// Fused distance-aware MHSA (flash-style, online softmax).
// One wave per (head, 16-query-row tile). q pre-scaled by 1/sqrt(dh).
// K/V tiles (32x64 bf16) staged via async-to-LDS; QK^T and P*V via WMMA;
// dmask MLP (4->3->4) added into C-frags in VALU. ctx written as bf16.
// ---------------------------------------------------------------------------
__global__ __launch_bounds__(32) void attn_k(
    const bf16_t* __restrict__ qb, const bf16_t* __restrict__ kb,
    const bf16_t* __restrict__ vb, const float* __restrict__ coords,
    const float* __restrict__ gdW1, const float* __restrict__ gdb1,
    const float* __restrict__ gdW2, const float* __restrict__ gdb2,
    bf16_t* __restrict__ ctxb)
{
  const int NT = N_NODES / 16;
  const int h = blockIdx.x / NT;
  const int n0 = (blockIdx.x % NT) << 4;
  const int lane = threadIdx.x;
  const int hi = lane >> 4, ln = lane & 15;

  __shared__ bf16_t Ks[32][64];   // [key][feat]  (head-h slice)
  __shared__ bf16_t Vs[32][64];   // [key][dim]
  __shared__ bf16_t P[16][36];

  // dmask MLP params (tiny; head-h slice of W2)
  float w1[12], b1v[3], w2r[3], b2v;
#pragma unroll
  for (int i = 0; i < 12; i++) w1[i] = gdW1[i];
#pragma unroll
  for (int i = 0; i < 3; i++) { b1v[i] = gdb1[i]; w2r[i] = gdW2[i * HEADS + h]; }
  b2v = gdb2[h];

  // Q fragments (K = 64 -> two 32-chunks)
  v16bf q0, q1;
#pragma unroll
  for (int e = 0; e < 16; e++) {
    int j = e >> 1, p = e & 1;
    int kk = ((j < 4) ? 0 : 16) + hi * 8 + (j & 3) * 2 + p;
    long base = (long)(n0 + ln) * DIM + h * 64;
    q0[e] = qb[base + kk];
    q1[e] = qb[base + 32 + kk];
  }

  float rx[8], ry[8], rz[8];
#pragma unroll
  for (int j = 0; j < 8; j++) {
    int n = n0 + j + 8 * hi;
    rx[j] = coords[n * 3 + 0]; ry[j] = coords[n * 3 + 1]; rz[j] = coords[n * 3 + 2];
  }

  float rmax[8], rsum[8];
  v8f acc[4];
#pragma unroll
  for (int j = 0; j < 8; j++) { rmax[j] = -1e30f; rsum[j] = 0.0f; }
#pragma unroll
  for (int i = 0; i < 4; i++)
#pragma unroll
    for (int j = 0; j < 8; j++) acc[i][j] = 0.0f;

  for (int m0 = 0; m0 < N_NODES; m0 += 32) {
    // stage K and V tiles for this 32-key block (head-h 64-dim slices)
#pragma unroll
    for (int i = 0; i < 8; i++) {
      int chunk = i * 32 + lane;          // 256 chunks of 16B per tile
      int key = chunk >> 3, c8 = chunk & 7;
      long gb = (long)(m0 + key) * DIM + h * 64 + c8 * 8;
      async_b128(kb + gb, &Ks[key][c8 * 8]);
      async_b128(vb + gb, &Vs[key][c8 * 8]);
    }
    wait_async0();

    v8f s[2];
#pragma unroll
    for (int st = 0; st < 2; st++) {
#pragma unroll
      for (int j = 0; j < 8; j++) s[st][j] = 0.0f;
      const int key = m0 + st * 16 + ln;
      v16bf bk0, bk1;
#pragma unroll
      for (int e = 0; e < 16; e++) {
        int j = e >> 1, p = e & 1;
        int kk = ((j < 4) ? 0 : 16) + hi * 8 + (j & 3) * 2 + p;
        bk0[e] = Ks[st * 16 + ln][kk];
        bk1[e] = Ks[st * 16 + ln][32 + kk];
      }
      s[st] = wmma_bf16(q0, bk0, s[st]);
      s[st] = wmma_bf16(q1, bk1, s[st]);
      // distance-mask MLP for this lane's column
      float cx = coords[key * 3 + 0], cy = coords[key * 3 + 1], cz = coords[key * 3 + 2];
#pragma unroll
      for (int j = 0; j < 8; j++) {
        float dx = rx[j] - cx, dy = ry[j] - cy, dz = rz[j] - cz;
        float sq = dx * dx + dy * dy + dz * dz;
        float dist = (sq > 0.0f) ? sqrtf(sq) : 0.0f;
        float hm0 = fmaxf(dx * w1[0] + dy * w1[3] + dz * w1[6] + dist * w1[9]  + b1v[0], 0.0f);
        float hm1 = fmaxf(dx * w1[1] + dy * w1[4] + dz * w1[7] + dist * w1[10] + b1v[1], 0.0f);
        float hm2 = fmaxf(dx * w1[2] + dy * w1[5] + dz * w1[8] + dist * w1[11] + b1v[2], 0.0f);
        s[st][j] += hm0 * w2r[0] + hm1 * w2r[1] + hm2 * w2r[2] + b2v;
      }
    }
    // online softmax over the 32 keys of this tile
#pragma unroll
    for (int j = 0; j < 8; j++) {
      float v0 = fmaxf(s[0][j], s[1][j]);
#pragma unroll
      for (int off = 8; off >= 1; off >>= 1) v0 = fmaxf(v0, __shfl_xor(v0, off, 16));
      float nm = fmaxf(rmax[j], v0);
      float scale = __expf(rmax[j] - nm);
      float p0 = __expf(s[0][j] - nm);
      float p1 = __expf(s[1][j] - nm);
      s[0][j] = p0; s[1][j] = p1;
      float ps = p0 + p1;
#pragma unroll
      for (int off = 8; off >= 1; off >>= 1) ps += __shfl_xor(ps, off, 16);
      rsum[j] = rsum[j] * scale + ps;
      rmax[j] = nm;
#pragma unroll
      for (int ct = 0; ct < 4; ct++) acc[ct][j] *= scale;
    }
    // P (C-layout) -> LDS -> A-fragment
#pragma unroll
    for (int st = 0; st < 2; st++)
#pragma unroll
      for (int j = 0; j < 8; j++) P[j + 8 * hi][st * 16 + ln] = (bf16_t)s[st][j];
    __syncthreads();
    v16bf pf;
#pragma unroll
    for (int e = 0; e < 16; e++) {
      int j = e >> 1, p = e & 1;
      int kk = ((j < 4) ? 0 : 16) + hi * 8 + (j & 3) * 2 + p;
      pf[e] = P[ln][kk];
    }
#pragma unroll
    for (int ct = 0; ct < 4; ct++) {
      v16bf vf;
#pragma unroll
      for (int e = 0; e < 16; e++) {
        int j = e >> 1, p = e & 1;
        int kk = ((j < 4) ? 0 : 16) + hi * 8 + (j & 3) * 2 + p;    // key index within tile
        vf[e] = Vs[kk][ct * 16 + ln];
      }
      acc[ct] = wmma_bf16(pf, vf, acc[ct]);
    }
    __syncthreads();
  }

#pragma unroll
  for (int ct = 0; ct < 4; ct++)
#pragma unroll
    for (int j = 0; j < 8; j++) {
      int n = n0 + j + 8 * hi;
      ctxb[(long)n * DIM + h * 64 + ct * 16 + ln] = (bf16_t)(acc[ct][j] / rsum[j]);
    }
}

// ---------------------------------------------------------------------------
// Fused per-edge-head attention conv-MLP (128->128 relu ->64 softmax) + msg
// + segment-max into agg. 16 (edge,head) rows per wave; bf16 operands.
// ---------------------------------------------------------------------------
__global__ __launch_bounds__(128) void edgeatt_k(
    const bf16_t* __restrict__ pqn, const bf16_t* __restrict__ pvn,
    const bf16_t* __restrict__ edgep,
    const bf16_t* __restrict__ W1b, const bf16_t* __restrict__ W2b,
    const float* __restrict__ ab1, const float* __restrict__ ab2,
    const int* __restrict__ src, const int* __restrict__ dst,
    float* __restrict__ agg)
{
  const int lane = threadIdx.x & 31, wave = threadIdx.x >> 5;
  const int hi = lane >> 4, ln = lane & 15;
  const int r0 = (blockIdx.x * 4 + wave) << 4;

  __shared__ bf16_t Hs[4][16][132];
  __shared__ float  Os[4][16][68];

  const int r = r0 + ln;
  const int e = r >> 2, h = r & 3;
  const int se = src[e], de = dst[e];
  const bf16_t* qrow = pqn + (long)se * DIM;
  const bf16_t* erow = edgep + (long)e * DIM;

  // ---- layer 1: D[16,128] = X[16,128] * W1^T ----
  v8f c1[8];
#pragma unroll
  for (int i = 0; i < 8; i++)
#pragma unroll
    for (int j = 0; j < 8; j++) c1[i][j] = 0.0f;

  for (int kb = 0; kb < 4; kb++) {
    v16bf af;
#pragma unroll
    for (int el = 0; el < 16; el++) {
      int j = el >> 1, p = el & 1;
      int kk = ((j < 4) ? 0 : 16) + hi * 8 + (j & 3) * 2 + p;
      int cidx = kb * 32 + kk;
      af[el] = (cidx < 64) ? qrow[cidx * 4 + h] : erow[(cidx - 64) * 4 + h];
    }
#pragma unroll
    for (int ot = 0; ot < 8; ot++) {
      v16bf bfv;
      int o = ot * 16 + ln;
#pragma unroll
      for (int el = 0; el < 16; el++) {
        int j = el >> 1, p = el & 1;
        int kk2 = ((j < 4) ? 0 : 16) + hi * 8 + (j & 3) * 2 + p;
        bfv[el] = W1b[o * 128 + kb * 32 + kk2];
      }
      c1[ot] = wmma_bf16(af, bfv, c1[ot]);
    }
  }
#pragma unroll
  for (int ot = 0; ot < 8; ot++) {
    float bb = ab1[ot * 16 + ln];
#pragma unroll
    for (int j = 0; j < 8; j++) {
      float v = fmaxf(c1[ot][j] + bb, 0.0f);
      Hs[wave][j + 8 * hi][ot * 16 + ln] = (bf16_t)v;
    }
  }
  __syncthreads();

  // ---- layer 2: D[16,64] = H[16,128] * W2^T ----
  v8f c2[4];
#pragma unroll
  for (int i = 0; i < 4; i++)
#pragma unroll
    for (int j = 0; j < 8; j++) c2[i][j] = 0.0f;

  for (int kb = 0; kb < 4; kb++) {
    v16bf af2;
#pragma unroll
    for (int el = 0; el < 16; el++) {
      int j = el >> 1, p = el & 1;
      int kk = ((j < 4) ? 0 : 16) + hi * 8 + (j & 3) * 2 + p;
      af2[el] = Hs[wave][ln][kb * 32 + kk];
    }
#pragma unroll
    for (int ot = 0; ot < 4; ot++) {
      v16bf bf2;
      int o = ot * 16 + ln;
#pragma unroll
      for (int el = 0; el < 16; el++) {
        int j = el >> 1, p = el & 1;
        int kk2 = ((j < 4) ? 0 : 16) + hi * 8 + (j & 3) * 2 + p;
        bf2[el] = W2b[o * 128 + kb * 32 + kk2];
      }
      c2[ot] = wmma_bf16(af2, bf2, c2[ot]);
    }
  }
#pragma unroll
  for (int ot = 0; ot < 4; ot++) {
    float bb = ab2[ot * 16 + ln];
#pragma unroll
    for (int j = 0; j < 8; j++) Os[wave][j + 8 * hi][ot * 16 + ln] = c2[ot][j] + bb;
  }
  __syncthreads();

  // softmax over 64 outputs (lane handles row ln, cols [hi*32, hi*32+32))
  float lm = -1e30f;
#pragma unroll
  for (int o = 0; o < 32; o++) lm = fmaxf(lm, Os[wave][ln][hi * 32 + o]);
  lm = fmaxf(lm, __shfl_xor(lm, 16, 32));
  float pvv[32], ls = 0.0f;
#pragma unroll
  for (int o = 0; o < 32; o++) { pvv[o] = __expf(Os[wave][ln][hi * 32 + o] - lm); ls += pvv[o]; }
  ls += __shfl_xor(ls, 16, 32);
  float inv = 1.0f / ls;
  const bf16_t* vrow = pvn + (long)de * DIM;
#pragma unroll
  for (int o = 0; o < 32; o++) {
    int col = (hi * 32 + o) * 4 + h;
    float msg = pvv[o] * inv * (float)vrow[col];
    atomicMaxF(&agg[(long)se * DIM + col], msg);
  }
}

// ---------------------------------------------------------------------------
// small elementwise / scatter kernels
// ---------------------------------------------------------------------------
__global__ void fill_f(float* p, float v, long n) {
  for (long i = blockIdx.x * (long)blockDim.x + threadIdx.x; i < n; i += (long)gridDim.x * blockDim.x) p[i] = v;
}
__global__ void fill_i(int* p, int v, long n) {
  for (long i = blockIdx.x * (long)blockDim.x + threadIdx.x; i < n; i += (long)gridDim.x * blockDim.x) p[i] = v;
}
__global__ void cvt_bf16_k(const float* s, bf16_t* d, long n) {
  for (long i = blockIdx.x * (long)blockDim.x + threadIdx.x; i < n; i += (long)gridDim.x * blockDim.x)
    d[i] = (bf16_t)s[i];
}
__global__ void edge_cnt_k(const int* src, const int* dst, float* cs, float* cd) {
  int e = blockIdx.x * blockDim.x + threadIdx.x;
  if (e < N_EDGES) { atomicAdd(&cs[src[e]], 1.0f); atomicAdd(&cd[dst[e]], 1.0f); }
}
__global__ void edge_sum_k(const float* ef, const int* src, const int* dst, float* ssum, float* osum) {
  long i = blockIdx.x * (long)blockDim.x + threadIdx.x;
  if (i < (long)N_EDGES * DIM) {
    int e = (int)(i >> 8), d = (int)(i & 255);
    float v = ef[i];
    atomicAdd(&ssum[(long)src[e] * DIM + d], v);
    atomicAdd(&osum[(long)dst[e] * DIM + d], v);
  }
}
__global__ void seg_div_k(float* ssum, float* osum, const float* cs, const float* cd,
                          bf16_t* ssb, bf16_t* osb) {
  long i = blockIdx.x * (long)blockDim.x + threadIdx.x;
  if (i < (long)N_NODES * DIM) {
    int n = (int)(i >> 8);
    float a = ssum[i] / fmaxf(cs[n], 1.0f);
    float b = osum[i] / fmaxf(cd[n], 1.0f);
    ssum[i] = a; osum[i] = b;
    ssb[i] = (bf16_t)a; osb[i] = (bf16_t)b;
  }
}
__global__ void node_out_k(const float* mhsa, const float* eatt, float* out) {
  long i = blockIdx.x * (long)blockDim.x + threadIdx.x;
  if (i < (long)N_NODES * DIM) out[i] = fmaxf(mhsa[i], 0.0f) * eatt[i];
}
__global__ void table_scatter_k(const int* src, const int* dst, int* table) {
  int e = blockIdx.x * blockDim.x + threadIdx.x;
  if (e < N_EDGES) table[(long)src[e] * N_NODES + dst[e]] = e + 1;
}
__global__ void rev_gather_k(const int* src, const int* dst, const int* table, int* revIdx) {
  int e = blockIdx.x * blockDim.x + threadIdx.x;
  if (e < N_EDGES) revIdx[e] = table[(long)dst[e] * N_NODES + src[e]] - 1;
}
__global__ void agg_fix_k(float* agg, bf16_t* aggb) {
  long i = blockIdx.x * (long)blockDim.x + threadIdx.x;
  if (i < (long)N_NODES * DIM) {
    float v = agg[i];
    if (!__builtin_isfinite(v)) v = 0.0f;
    agg[i] = v;
    aggb[i] = (bf16_t)v;
  }
}

// ---------------------------------------------------------------------------
static void gemm(hipStream_t s, const bf16_t* A, int lda, const int* idx, const bf16_t* zp,
                 const bf16_t* W, int ldw, const float* bias,
                 int accum, int writeF, int act, float oscale,
                 float* Cf, bf16_t* Cb, int ldc, int M, int N, int K) {
  gemm_k<<<dim3((M / 64) * (N / 64)), 128, 0, s>>>(A, lda, idx, zp, W, ldw, bias,
                                                   accum, writeF, act, oscale, Cf, Cb, ldc, M, N, K);
}
static void cvt(hipStream_t s, const float* src, bf16_t* dst, long n) {
  int blocks = (int)((n + 255) / 256); if (blocks > 2048) blocks = 2048;
  cvt_bf16_k<<<blocks, 256, 0, s>>>(src, dst, n);
}

extern "C" void kernel_launch(void* const* d_in, const int* in_sizes, int n_in,
                              void* d_out, int out_size, void* d_ws, size_t ws_size,
                              hipStream_t stream) {
  (void)in_sizes; (void)n_in; (void)out_size; (void)ws_size;
  const float* node_f = (const float*)d_in[0];
  const float* edge_f = (const float*)d_in[1];
  const float* coords = (const float*)d_in[2];
  const int*   eidx   = (const int*)d_in[3];
  const int*   src    = eidx;
  const int*   dst    = eidx + N_EDGES;
  const float *Wq = (const float*)d_in[5],  *bq = (const float*)d_in[6];
  const float *Wk = (const float*)d_in[7],  *bk = (const float*)d_in[8];
  const float *Wv = (const float*)d_in[9],  *bv = (const float*)d_in[10];
  const float *Wo = (const float*)d_in[11], *bo = (const float*)d_in[12];
  const float *gdW1 = (const float*)d_in[13], *gdb1 = (const float*)d_in[14];
  const float *gdW2 = (const float*)d_in[15], *gdb2 = (const float*)d_in[16];
  const float *twW1 = (const float*)d_in[17], *twb1 = (const float*)d_in[18];
  const float *twW2 = (const float*)d_in[19], *twb2 = (const float*)d_in[20];
  const float *neW1 = (const float*)d_in[21], *neb1 = (const float*)d_in[22];
  const float *neW2 = (const float*)d_in[23], *neb2 = (const float*)d_in[24];
  const float *atW1 = (const float*)d_in[25], *atb1 = (const float*)d_in[26];
  const float *atW2 = (const float*)d_in[27], *atb2 = (const float*)d_in[28];
  const float *peW = (const float*)d_in[29], *peb = (const float*)d_in[30];
  const float *pqW = (const float*)d_in[31], *pqb = (const float*)d_in[32];
  const float *pvW = (const float*)d_in[33], *pvb = (const float*)d_in[34];
  const float *prW1 = (const float*)d_in[35], *prb1 = (const float*)d_in[36];
  const float *prW2 = (const float*)d_in[37], *prb2 = (const float*)d_in[38];

  float* out_node = (float*)d_out;                                   // [N,256]
  float* out_edge = out_node + (long)N_NODES * DIM;                  // [E,256]
  float* out_xx   = out_edge + (long)N_EDGES * DIM;                  // [N,256]

  // ---- workspace bump allocator ----
  char* wp = (char*)d_ws;
  auto alloc = [&](size_t bytes) { void* p = (void*)wp; wp += (bytes + 255) & ~(size_t)255; return p; };
  const long NN = (long)N_NODES * DIM, NE = (long)N_EDGES * DIM;

  bf16_t* node_b = (bf16_t*)alloc(NN * 2);
  bf16_t* edge_b = (bf16_t*)alloc(NE * 2);
  bf16_t* zp     = (bf16_t*)alloc(64);               // zero page for idx<0
  // bf16 weights
  bf16_t *Wq_b = (bf16_t*)alloc(65536 * 2), *Wk_b = (bf16_t*)alloc(65536 * 2);
  bf16_t *Wv_b = (bf16_t*)alloc(65536 * 2), *Wo_b = (bf16_t*)alloc(65536 * 2);
  bf16_t *twW1_b = (bf16_t*)alloc(131072 * 2), *twW2_b = (bf16_t*)alloc(65536 * 2);
  bf16_t *neW1_b = (bf16_t*)alloc(524288 * 2), *neW2_b = (bf16_t*)alloc(131072 * 2);
  bf16_t *peW_b = (bf16_t*)alloc(65536 * 2), *pqW_b = (bf16_t*)alloc(65536 * 2);
  bf16_t *pvW_b = (bf16_t*)alloc(65536 * 2);
  bf16_t *prW1_b = (bf16_t*)alloc(262144 * 2), *prW2_b = (bf16_t*)alloc(131072 * 2);
  bf16_t *W1b = (bf16_t*)alloc(16384 * 2), *W2b = (bf16_t*)alloc(8192 * 2);
  // activations
  bf16_t* qb    = (bf16_t*)alloc(NN * 2);
  bf16_t* kbuf  = (bf16_t*)alloc(NN * 2);
  bf16_t* vbuf  = (bf16_t*)alloc(NN * 2);
  bf16_t* ctx_b = (bf16_t*)alloc(NN * 2);
  float*  mhsa  = (float*)alloc(NN * 4);
  float*  cs    = (float*)alloc((size_t)N_NODES * 4);
  float*  cd    = (float*)alloc((size_t)N_NODES * 4);
  float*  ssum  = (float*)alloc(NN * 4);
  float*  osum  = (float*)alloc(NN * 4);
  bf16_t* ssb   = (bf16_t*)alloc(NN * 2);
  bf16_t* osb   = (bf16_t*)alloc(NN * 2);
  float*  twh_f = (float*)alloc(NN * 4);
  bf16_t* twh_b = (bf16_t*)alloc(NN * 2);
  float*  eatt  = (float*)alloc(NN * 4);
  int*    table = (int*)alloc((size_t)N_NODES * N_NODES * 4);
  int*    revIdx= (int*)alloc((size_t)N_EDGES * 4);
  float*  neh_f = (float*)alloc((size_t)N_EDGES * 512 * 4);
  bf16_t* neh_b = (bf16_t*)alloc((size_t)N_EDGES * 512 * 2);
  bf16_t* pqn_b = (bf16_t*)alloc(NN * 2);
  bf16_t* pvn_b = (bf16_t*)alloc(NN * 2);
  bf16_t* edp_b = (bf16_t*)alloc(NE * 2);
  float*  agg   = (float*)alloc(NN * 4);
  bf16_t* agg_b = (bf16_t*)alloc(NN * 2);
  float*  prh_f = (float*)alloc((size_t)N_NODES * 512 * 4);
  bf16_t* prh_b = (bf16_t*)alloc((size_t)N_NODES * 512 * 2);

  // ===== 0. bf16 conversions =====
  cvt(stream, node_f, node_b, NN);
  cvt(stream, edge_f, edge_b, NE);
  fill_f<<<1, 64, 0, stream>>>((float*)zp, 0.0f, 16);
  cvt(stream, Wq, Wq_b, 65536);   cvt(stream, Wk, Wk_b, 65536);
  cvt(stream, Wv, Wv_b, 65536);   cvt(stream, Wo, Wo_b, 65536);
  cvt(stream, twW1, twW1_b, 131072); cvt(stream, twW2, twW2_b, 65536);
  cvt(stream, neW1, neW1_b, 524288); cvt(stream, neW2, neW2_b, 131072);
  cvt(stream, peW, peW_b, 65536); cvt(stream, pqW, pqW_b, 65536);
  cvt(stream, pvW, pvW_b, 65536);
  cvt(stream, prW1, prW1_b, 262144); cvt(stream, prW2, prW2_b, 131072);
  cvt(stream, atW1, W1b, 16384);  cvt(stream, atW2, W2b, 8192);

  // ===== 1. MHSA path =====
  gemm(stream, node_b, DIM, nullptr, zp, Wq_b, DIM, bq, 0, 0, 0, 0.125f, nullptr, qb, DIM, N_NODES, DIM, DIM);
  gemm(stream, node_b, DIM, nullptr, zp, Wk_b, DIM, bk, 0, 0, 0, 1.0f, nullptr, kbuf, DIM, N_NODES, DIM, DIM);
  gemm(stream, node_b, DIM, nullptr, zp, Wv_b, DIM, bv, 0, 0, 0, 1.0f, nullptr, vbuf, DIM, N_NODES, DIM, DIM);
  attn_k<<<dim3(HEADS * (N_NODES / 16)), 32, 0, stream>>>(qb, kbuf, vbuf, coords, gdW1, gdb1, gdW2, gdb2, ctx_b);
  gemm(stream, ctx_b, DIM, nullptr, zp, Wo_b, DIM, bo, 0, 1, 0, 1.0f, mhsa, nullptr, DIM, N_NODES, DIM, DIM);

  // ===== 2. Twinning edge attention =====
  fill_f<<<256, 256, 0, stream>>>(cs, 0.0f, N_NODES);
  fill_f<<<256, 256, 0, stream>>>(cd, 0.0f, N_NODES);
  fill_f<<<1024, 256, 0, stream>>>(ssum, 0.0f, NN);
  fill_f<<<1024, 256, 0, stream>>>(osum, 0.0f, NN);
  edge_cnt_k<<<N_EDGES / 256, 256, 0, stream>>>(src, dst, cs, cd);
  edge_sum_k<<<(unsigned)(NE / 256), 256, 0, stream>>>(edge_f, src, dst, ssum, osum);
  seg_div_k<<<(unsigned)(NN / 256), 256, 0, stream>>>(ssum, osum, cs, cd, ssb, osb);
  gemm(stream, ssb, DIM, nullptr, zp, twW1_b, DIM, nullptr, 0, 1, 0, 1.0f, twh_f, nullptr, DIM, N_NODES, DIM, DIM);
  gemm(stream, osb, DIM, nullptr, zp, twW1_b + 256 * DIM, DIM, twb1, 1, 0, 1, 1.0f, twh_f, twh_b, DIM, N_NODES, DIM, DIM);
  gemm(stream, twh_b, DIM, nullptr, zp, twW2_b, DIM, twb2, 0, 1, 2, 1.0f, eatt, nullptr, DIM, N_NODES, DIM, DIM);
  node_out_k<<<(unsigned)(NN / 256), 256, 0, stream>>>(mhsa, eatt, out_node);

  // ===== 3. reverse-edge lookup =====
  fill_i<<<4096, 256, 0, stream>>>(table, 0, (long)N_NODES * N_NODES);
  table_scatter_k<<<N_EDGES / 256, 256, 0, stream>>>(src, dst, table);
  rev_gather_k<<<N_EDGES / 256, 256, 0, stream>>>(src, dst, table, revIdx);

  // ===== 4. edge MLP (split concat into 4 accumulating GEMMs) =====
  gemm(stream, node_b, DIM, src,    zp, neW1_b,             512, nullptr, 0, 1, 0, 1.0f, neh_f, nullptr, 512, N_EDGES, 512, DIM);
  gemm(stream, edge_b, DIM, nullptr,zp, neW1_b + 256 * 512, 512, nullptr, 1, 1, 0, 1.0f, neh_f, nullptr, 512, N_EDGES, 512, DIM);
  gemm(stream, edge_b, DIM, revIdx, zp, neW1_b + 512 * 512, 512, nullptr, 1, 1, 0, 1.0f, neh_f, nullptr, 512, N_EDGES, 512, DIM);
  gemm(stream, node_b, DIM, dst,    zp, neW1_b + 768 * 512, 512, neb1,    1, 0, 1, 1.0f, neh_f, neh_b, 512, N_EDGES, 512, DIM);
  gemm(stream, neh_b, 512, nullptr, zp, neW2_b, DIM, neb2, 0, 1, 0, 1.0f, out_edge, nullptr, DIM, N_EDGES, DIM, 512);

  // ===== 5. per-edge attention aggregation =====
  gemm(stream, node_b, DIM, nullptr, zp, pqW_b, DIM, pqb, 0, 0, 0, 1.0f, nullptr, pqn_b, DIM, N_NODES, DIM, DIM);
  gemm(stream, node_b, DIM, nullptr, zp, pvW_b, DIM, pvb, 0, 0, 0, 1.0f, nullptr, pvn_b, DIM, N_NODES, DIM, DIM);
  gemm(stream, edge_b, DIM, nullptr, zp, peW_b, DIM, peb, 0, 0, 0, 1.0f, nullptr, edp_b, DIM, N_EDGES, DIM, DIM);
  fill_f<<<1024, 256, 0, stream>>>(agg, -INFINITY, NN);
  edgeatt_k<<<dim3((N_EDGES * HEADS) / 64), 128, 0, stream>>>(pqn_b, pvn_b, edp_b, W1b, W2b, atb1, atb2, src, dst, agg);
  agg_fix_k<<<(unsigned)(NN / 256), 256, 0, stream>>>(agg, agg_b);

  // ===== 6. projection MLP =====
  gemm(stream, node_b, DIM, nullptr, zp, prW1_b,             512, nullptr, 0, 1, 0, 1.0f, prh_f, nullptr, 512, N_NODES, 512, DIM);
  gemm(stream, agg_b,  DIM, nullptr, zp, prW1_b + 256 * 512, 512, prb1,    1, 0, 1, 1.0f, prh_f, prh_b, 512, N_NODES, 512, DIM);
  gemm(stream, prh_b, 512, nullptr, zp, prW2_b, DIM, prb2, 0, 1, 0, 1.0f, out_xx, nullptr, DIM, N_NODES, DIM, 512);
}
